// SpatialTransformerPredictor_19533511262829
// MI455X (gfx1250) — matrix operand order repaired and checked by reading, not compiled, not run on hardware
//
#include <hip/hip_runtime.h>
#include <cstdint>
#include <cstddef>

typedef __attribute__((ext_vector_type(16))) _Float16 v16h;
typedef __attribute__((ext_vector_type(8)))  float    v8f;
typedef _Float16 h16;

// ---------------------------------------------------------------
// CDNA5 async global->LDS staging (guarded; sync fallback)
// ---------------------------------------------------------------
#if defined(__HIP_DEVICE_COMPILE__)
#if __has_builtin(__builtin_amdgcn_global_load_async_to_lds_b128) && __has_builtin(__builtin_amdgcn_s_wait_asynccnt)
#define USE_ASYNC_LDS 1
#endif
#endif

#if defined(USE_ASYNC_LDS)
typedef int v4i_ __attribute__((vector_size(16)));
typedef __attribute__((address_space(1))) v4i_* as1_v4i;
typedef __attribute__((address_space(3))) v4i_* as3_v4i;
#endif

__device__ __forceinline__ void cp16_to_lds(const h16* g, h16* l, bool valid){
#if defined(USE_ASYNC_LDS)
  if (valid) {
    __builtin_amdgcn_global_load_async_to_lds_b128(
        (as1_v4i)(unsigned long long)(const void*)g,
        (as3_v4i)(unsigned long long)(const void*)l, 0, 0);
  } else {
    uint4 z = {0,0,0,0};
    *(uint4*)l = z;
  }
#else
  uint4 v = {0,0,0,0};
  if (valid) v = *(const uint4*)g;
  *(uint4*)l = v;
#endif
}
__device__ __forceinline__ void stage_fence(){
#if defined(USE_ASYNC_LDS)
  __builtin_amdgcn_s_wait_asynccnt(0);
#endif
  __syncthreads();
}

// ======================================================================
// Utility kernels
// ======================================================================
__global__ __launch_bounds__(256) void k_cvt_h(const float* __restrict__ s, h16* __restrict__ d, int n){
  int i = blockIdx.x*256 + threadIdx.x;
  if (i < n) d[i] = (h16)s[i];
}
// linear weight fp32 [K][N] -> fp16 transposed [N][K]
__global__ __launch_bounds__(256) void k_cvt_lin_t(const float* __restrict__ s, h16* __restrict__ d, int K, int N){
  int i = blockIdx.x*256 + threadIdx.x;
  if (i >= K*N) return;
  int n = i % N, k = i / N;
  d[(size_t)n*K + k] = (h16)s[i];
}
__global__ __launch_bounds__(256) void k_copy_f(const float* __restrict__ s, float* __restrict__ d, int n){
  int i = blockIdx.x*256 + threadIdx.x;
  if (i < n) d[i] = s[i];
}
__global__ __launch_bounds__(256) void k_add_f(float* __restrict__ x, const float* __restrict__ y, int n){
  int i = blockIdx.x*256 + threadIdx.x;
  if (i < n) x[i] += y[i];
}

// Build fp16 NHWC(C=1) input for 44 images: [0..3]=target, [4..43]=objects
__global__ __launch_bounds__(256) void k_build_input(const float* __restrict__ tgt,
                                                     const float* __restrict__ obj,
                                                     h16* __restrict__ out, int total){
  int i = blockIdx.x*256 + threadIdx.x;
  if (i >= total) return;
  const int hw = 224*224;
  int img = i / hw, p = i - img*hw;
  float v = (img < 4) ? tgt[(size_t)img*hw + p] : obj[(size_t)(img-4)*hw + p];
  out[i] = (h16)v;
}

// conv1 weight fold: (64,3,7,7) OIHW fp32 -> [K=49][64] fp16 summed over Cin
__global__ __launch_bounds__(256) void k_fold_conv1(const float* __restrict__ w, h16* __restrict__ dst){
  int i = blockIdx.x*256 + threadIdx.x;
  if (i >= 49*64) return;
  int k = i / 64, o = i - (i/64)*64;
  int r = k / 7, s = k - r*7;
  float v = 0.f;
  for (int c = 0; c < 3; ++c) v += w[(((size_t)o*3 + c)*7 + r)*7 + s];
  dst[(size_t)k*64 + o] = (h16)v;
}

// conv weight OIHW fp32 -> TRANSPOSED fp16 [Cout][K], K=(r*S+s)*Cin+c
__global__ __launch_bounds__(256) void k_cvt_convw_t(const float* __restrict__ w, h16* __restrict__ dst,
                                                     int Cout,int Cin,int R,int S){
  int tot = Cout*Cin*R*S;
  int i = blockIdx.x*256 + threadIdx.x;
  if (i >= tot) return;
  int s = i % S; int t = i / S;
  int r = t % R; t /= R;
  int c = t % Cin; int o = t / Cin;
  dst[(size_t)o*(R*S*Cin) + (size_t)((r*S + s)*Cin + c)] = (h16)w[i];
}

// ======================================================================
// Fast implicit-GEMM conv for Cin % 32 == 0 (all ResNet stage convs).
// Tile 128(M) x 64(N), K-step 32. Weights pre-transposed [Cout][K].
// Double-buffered LDS in WMMA fragment-major order: async b128 staging
// of step k+1 overlaps the WMMAs of step k; one fence per K-step.
// Tap (r,s,cbase) is maintained incrementally (no divisions in loop).
// ======================================================================
__global__ __launch_bounds__(256) void k_conv_wmma_c32(
    const h16* __restrict__ in, const h16* __restrict__ wgtT,
    const float* __restrict__ bns, const float* __restrict__ bnb,
    const h16* res, h16* out,
    int Hin,int Win,int Cin,int Hout,int Wout,int Cout,
    int R,int S,int stride,int pad,int Mtot,int relu)
{
  __shared__ __attribute__((aligned(32))) h16 AsF[2][8][32][16];
  __shared__ __attribute__((aligned(32))) h16 BsF[2][4][32][16];
  const int tid  = threadIdx.x;
  const int lane = tid & 31;
  const int wid  = tid >> 5;
  const int wm   = wid & 3;            // 4 x 32 M rows
  const int wn   = wid >> 2;           // 2 x 32 N cols
  const int lrow = lane & 15;
  const int lhi  = lane >> 4;
  const int m0 = blockIdx.x * 128;
  const int n0 = blockIdx.y * 64;
  const int Ktot = R*S*Cin;
  const int HWo  = Hout*Wout;

  // A staging: 512 chunks of 8 halves; 2 chunks/thread
  const int arow0 = tid >> 2;          // 0..63
  const int akc   = tid & 3;           // K chunk 0..3
  int aimg[2], aihb[2], aiwb[2]; bool amv[2];
  #pragma unroll
  for (int q = 0; q < 2; ++q){
    int m = m0 + arow0 + q*64;
    amv[q] = (m < Mtot);
    int img = m / HWo;
    int rem = m - img*HWo;
    int oh  = rem / Wout;
    int ow  = rem - oh*Wout;
    aimg[q] = img;
    aihb[q] = oh*stride - pad;
    aiwb[q] = ow*stride - pad;
  }
  const int a_lhi = (akc & 1) ? 16 : 0;
  const int a_j   = (akc >> 1) * 8;
  const int bnn = tid >> 2;            // 0..63
  const int bkc = tid & 3;
  const int b_lane = (bnn & 15) + ((bkc & 1) ? 16 : 0);
  const int b_j    = (bkc >> 1) * 8;
  const int b_sub  = bnn >> 4;

  auto stageA = [&](int buf, int r, int s, int cb){
    #pragma unroll
    for (int q = 0; q < 2; ++q){
      int ih = aihb[q] + r, iw = aiwb[q] + s;
      bool v = amv[q] && ih >= 0 && ih < Hin && iw >= 0 && iw < Win;
      const h16* g = in + (((size_t)aimg[q]*Hin + ih)*Win + iw)*Cin + cb + akc*8;
      cp16_to_lds(g, &AsF[buf][(arow0 >> 4) + q*4][(arow0 & 15) + a_lhi][a_j], v);
    }
  };
  auto stageB = [&](int buf, int k0){
    cp16_to_lds(wgtT + (size_t)(n0 + bnn)*Ktot + k0 + bkc*8,
                &BsF[buf][b_sub][b_lane][b_j], true);
    if (k0 + 32 < Ktot)
      __builtin_prefetch(wgtT + (size_t)(n0 + bnn)*Ktot + k0 + 32 + bkc*8, 0, 3);
  };

  v8f acc[2][2] = {};

  // prologue: stage first K chunk into buffer 0
  int rr = 0, ss = 0, cb = 0;
  stageA(0, 0, 0, 0);
  stageB(0, 0);

  for (int k0 = 0; k0 < Ktot; k0 += 32) {
    const int cur = (k0 >> 5) & 1;
    stage_fence();                       // buf[cur] ready; prior reads done
    if (k0 + 32 < Ktot) {                // issue async for next chunk
      cb += 32;
      if (cb >= Cin) { cb = 0; if (++ss == S) { ss = 0; ++rr; } }
      stageA(cur ^ 1, rr, ss, cb);
      stageB(cur ^ 1, k0 + 32);
    }
    v16h a0 = *(const v16h*)&AsF[cur][wm*2 + 0][lane][0];
    v16h a1 = *(const v16h*)&AsF[cur][wm*2 + 1][lane][0];
    v16h b0 = *(const v16h*)&BsF[cur][wn*2 + 0][lane][0];
    v16h b1 = *(const v16h*)&BsF[cur][wn*2 + 1][lane][0];
    acc[0][0] = __builtin_amdgcn_wmma_f32_16x16x32_f16(false, a0, false, b0, (short)0, acc[0][0], false, false);
    acc[0][1] = __builtin_amdgcn_wmma_f32_16x16x32_f16(false, a0, false, b1, (short)0, acc[0][1], false, false);
    acc[1][0] = __builtin_amdgcn_wmma_f32_16x16x32_f16(false, a1, false, b0, (short)0, acc[1][0], false, false);
    acc[1][1] = __builtin_amdgcn_wmma_f32_16x16x32_f16(false, a1, false, b1, (short)0, acc[1][1], false, false);
  }

  #pragma unroll
  for (int st2 = 0; st2 < 2; ++st2){
    #pragma unroll
    for (int r8 = 0; r8 < 8; ++r8){
      int m = m0 + wm*32 + st2*16 + lhi*8 + r8;
      if (m >= Mtot) continue;
      #pragma unroll
      for (int f = 0; f < 2; ++f){
        int n = n0 + wn*32 + f*16 + lrow;
        float v = acc[st2][f][r8];
        v = v * bns[n] + bnb[n];
        if (res)  v += (float)res[(size_t)m*Cout + n];
        if (relu) v = fmaxf(v, 0.f);
        out[(size_t)m*Cout + n] = (h16)v;
      }
    }
  }
}

// ======================================================================
// Templated conv (compile-time R,S,Cin) for conv1 (7x7, Cin=1)
// ======================================================================
template<int R, int S, int CIN>
__global__ __launch_bounds__(256) void k_conv_gen(
    const h16* __restrict__ in, const h16* __restrict__ wgt,
    const float* __restrict__ bns, const float* __restrict__ bnb,
    const h16* res, h16* out,
    int Hin,int Win,int Hout,int Wout,int Cout,
    int stride,int pad,int Mtot,int relu)
{
  __shared__ h16 As[64][40];
  __shared__ h16 Bs[32][72];
  const int KT = R*S*CIN;
  const int tid  = threadIdx.x;
  const int lane = tid & 31;
  const int wid  = tid >> 5;
  const int wm   = wid & 3;
  const int wn   = wid >> 2;
  const int lrow = lane & 15;
  const int lhi  = lane >> 4;
  const int m0 = blockIdx.x * 64;
  const int n0 = blockIdx.y * 64;
  const int HWo = Hout*Wout;

  const int ar  = tid >> 2;
  const int kc0 = (tid & 3) * 8;
  const int am  = m0 + ar;
  int img = am / HWo;
  int rem = am - img*HWo;
  int oh  = rem / Wout;
  int ow  = rem - oh*Wout;
  const int ihb = oh*stride - pad;
  const int iwb = ow*stride - pad;
  const int br  = tid >> 3;
  const int nc0 = (tid & 7) * 8;

  v8f acc0 = {}; v8f acc1 = {};

  for (int k0 = 0; k0 < KT; k0 += 32) {
    #pragma unroll
    for (int j = 0; j < 8; ++j) {
      int k = k0 + kc0 + j;
      h16 v = (h16)0.f;
      if (am < Mtot && k < KT) {
        int c  = k % CIN;
        int rs = k / CIN;
        int s  = rs % S;
        int r  = rs / S;
        int ih = ihb + r, iw = iwb + s;
        if (ih >= 0 && ih < Hin && iw >= 0 && iw < Win)
          v = in[(((size_t)img*Hin + ih)*Win + iw)*CIN + c];
      }
      As[ar][kc0 + j] = v;
    }
    {
      int k = k0 + br;
      #pragma unroll
      for (int j = 0; j < 8; ++j) {
        int n = n0 + nc0 + j;
        Bs[br][nc0 + j] = (k < KT && n < Cout) ? wgt[(size_t)k*Cout + n] : (h16)0.f;
      }
    }
    __syncthreads();
    v16h a, b0, b1;
    #pragma unroll
    for (int j = 0; j < 16; ++j) {
      int kk = (j < 8 ? j : 8 + j) + lhi*8;
      a[j] = As[wm*16 + lrow][kk];
    }
    #pragma unroll
    for (int j = 0; j < 16; ++j) {
      int kk = (j < 8 ? j : 8 + j) + lhi*8;
      b0[j] = Bs[kk][wn*32 + lrow];
      b1[j] = Bs[kk][wn*32 + 16 + lrow];
    }
    acc0 = __builtin_amdgcn_wmma_f32_16x16x32_f16(false, a, false, b0, (short)0, acc0, false, false);
    acc1 = __builtin_amdgcn_wmma_f32_16x16x32_f16(false, a, false, b1, (short)0, acc1, false, false);
    __syncthreads();
  }

  #pragma unroll
  for (int r8 = 0; r8 < 8; ++r8) {
    int m = m0 + wm*16 + lhi*8 + r8;
    if (m >= Mtot) continue;
    #pragma unroll
    for (int f = 0; f < 2; ++f) {
      int n = n0 + wn*32 + f*16 + lrow;
      if (n >= Cout) continue;
      float v = (f ? acc1[r8] : acc0[r8]);
      v = v * bns[n] + bnb[n];
      if (res)  v += (float)res[(size_t)m*Cout + n];
      if (relu) v = fmaxf(v, 0.f);
      out[(size_t)m*Cout + n] = (h16)v;
    }
  }
}

// ======================================================================
// Fast GEMM, transposed weights: C[M,N] = A[M,K] * Bt[N,K]^T + bias
// K % 32 == 0. Tile 64x64, double-buffered frag-major LDS, async b128.
// ======================================================================
__global__ __launch_bounds__(256) void k_gemm_wmma_t(
    const h16* __restrict__ A, const h16* __restrict__ Bt,
    const float* bias, float* C, int ldc,
    int M, int Nn, int K, int relu)
{
  __shared__ __attribute__((aligned(32))) h16 AsF[2][4][32][16];
  __shared__ __attribute__((aligned(32))) h16 BsF[2][4][32][16];
  const int tid  = threadIdx.x;
  const int lane = tid & 31;
  const int wid  = tid >> 5;
  const int wm   = wid & 3;            // 4 x 16 M rows
  const int wn   = wid >> 2;           // 2 x 32 N cols
  const int lrow = lane & 15;
  const int lhi  = lane >> 4;
  const int m0 = blockIdx.x * 64;
  const int n0 = blockIdx.y * 64;

  const int arow = tid >> 2;           // 0..63
  const int akc  = tid & 3;
  const int a_lhi = (akc & 1) ? 16 : 0;
  const int a_j   = (akc >> 1) * 8;
  const bool amv  = (m0 + arow) < M;
  const int bnn = tid >> 2;
  const int bkc = tid & 3;
  const int b_lane = (bnn & 15) + ((bkc & 1) ? 16 : 0);
  const int b_j    = (bkc >> 1) * 8;
  const int b_sub  = bnn >> 4;
  const bool bnv   = (n0 + bnn) < Nn;

  auto stage = [&](int buf, int k0){
    cp16_to_lds(A + (size_t)(m0 + arow)*K + k0 + akc*8,
                &AsF[buf][arow >> 4][(arow & 15) + a_lhi][a_j], amv);
    cp16_to_lds(Bt + (size_t)(n0 + bnn)*K + k0 + bkc*8,
                &BsF[buf][b_sub][b_lane][b_j], bnv);
    if (k0 + 32 < K && bnv)
      __builtin_prefetch(Bt + (size_t)(n0 + bnn)*K + k0 + 32 + bkc*8, 0, 3);
  };

  v8f acc0 = {}; v8f acc1 = {};
  stage(0, 0);

  for (int k0 = 0; k0 < K; k0 += 32) {
    const int cur = (k0 >> 5) & 1;
    stage_fence();
    if (k0 + 32 < K) stage(cur ^ 1, k0 + 32);
    v16h a  = *(const v16h*)&AsF[cur][wm][lane][0];
    v16h b0 = *(const v16h*)&BsF[cur][wn*2 + 0][lane][0];
    v16h b1 = *(const v16h*)&BsF[cur][wn*2 + 1][lane][0];
    acc0 = __builtin_amdgcn_wmma_f32_16x16x32_f16(false, a, false, b0, (short)0, acc0, false, false);
    acc1 = __builtin_amdgcn_wmma_f32_16x16x32_f16(false, a, false, b1, (short)0, acc1, false, false);
  }

  #pragma unroll
  for (int r8 = 0; r8 < 8; ++r8) {
    int m = m0 + wm*16 + lhi*8 + r8;
    if (m >= M) continue;
    #pragma unroll
    for (int f = 0; f < 2; ++f) {
      int n = n0 + wn*32 + f*16 + lrow;
      if (n >= Nn) continue;
      float v = (f ? acc1[r8] : acc0[r8]);
      if (bias) v += bias[n];
      if (relu) v = fmaxf(v, 0.f);
      C[(size_t)m*ldc + n] = v;
    }
  }
}

// ======================================================================
// Generic GEMM (non-transposed B, any K) -- only used for K=20 rel MLP
// ======================================================================
__global__ __launch_bounds__(256) void k_gemm_wmma(
    const h16* __restrict__ A, const h16* __restrict__ Bm, int ldb,
    const float* bias, float* C, int ldc,
    int M, int Nn, int K, int relu)
{
  __shared__ h16 As[64][40];
  __shared__ h16 Bs[32][72];
  const int tid  = threadIdx.x;
  const int lane = tid & 31;
  const int wid  = tid >> 5;
  const int wm   = wid & 3;
  const int wn   = wid >> 2;
  const int lrow = lane & 15;
  const int lhi  = lane >> 4;
  const int m0 = blockIdx.x * 64;
  const int n0 = blockIdx.y * 64;

  const int ar  = tid >> 2;
  const int kc0 = (tid & 3) * 8;
  const int am  = m0 + ar;
  const int br  = tid >> 3;
  const int nc0 = (tid & 7) * 8;

  v8f acc0 = {}; v8f acc1 = {};

  for (int k0 = 0; k0 < K; k0 += 32) {
    #pragma unroll
    for (int j = 0; j < 8; ++j) {
      int k = k0 + kc0 + j;
      As[ar][kc0 + j] = (am < M && k < K) ? A[(size_t)am*K + k] : (h16)0.f;
    }
    {
      int k = k0 + br;
      #pragma unroll
      for (int j = 0; j < 8; ++j) {
        int n = n0 + nc0 + j;
        Bs[br][nc0 + j] = (k < K && n < Nn) ? Bm[(size_t)k*ldb + n] : (h16)0.f;
      }
    }
    __syncthreads();
    v16h a, b0, b1;
    #pragma unroll
    for (int j = 0; j < 16; ++j) {
      int kk = (j < 8 ? j : 8 + j) + lhi*8;
      a[j] = As[wm*16 + lrow][kk];
    }
    #pragma unroll
    for (int j = 0; j < 16; ++j) {
      int kk = (j < 8 ? j : 8 + j) + lhi*8;
      b0[j] = Bs[kk][wn*32 + lrow];
      b1[j] = Bs[kk][wn*32 + 16 + lrow];
    }
    acc0 = __builtin_amdgcn_wmma_f32_16x16x32_f16(false, a, false, b0, (short)0, acc0, false, false);
    acc1 = __builtin_amdgcn_wmma_f32_16x16x32_f16(false, a, false, b1, (short)0, acc1, false, false);
    __syncthreads();
  }

  #pragma unroll
  for (int r8 = 0; r8 < 8; ++r8) {
    int m = m0 + wm*16 + lhi*8 + r8;
    if (m >= M) continue;
    #pragma unroll
    for (int f = 0; f < 2; ++f) {
      int n = n0 + wn*32 + f*16 + lrow;
      if (n >= Nn) continue;
      float v = (f ? acc1[r8] : acc0[r8]);
      if (bias) v += bias[n];
      if (relu) v = fmaxf(v, 0.f);
      C[(size_t)m*ldc + n] = v;
    }
  }
}

// ======================================================================
// Pooling
// ======================================================================
__global__ __launch_bounds__(256) void k_maxpool(const h16* __restrict__ in, h16* __restrict__ out,
                                                 int C,int Hin,int Win,int Hout,int Wout,int total){
  int i = blockIdx.x*256 + threadIdx.x;
  if (i >= total) return;
  int c = i % C; int t = i / C;
  int ow = t % Wout; t /= Wout;
  int oh = t % Hout; int img = t / Hout;
  float m = -3.0e38f;
  for (int kh = 0; kh < 3; ++kh){
    int ih = oh*2 - 1 + kh; if (ih < 0 || ih >= Hin) continue;
    for (int kw = 0; kw < 3; ++kw){
      int iw = ow*2 - 1 + kw; if (iw < 0 || iw >= Win) continue;
      m = fmaxf(m, (float)in[(((size_t)img*Hin + ih)*Win + iw)*C + c]);
    }
  }
  out[i] = (h16)m;
}
__global__ __launch_bounds__(256) void k_avgpool(const h16* __restrict__ in, h16* __restrict__ out, int n){
  int i = blockIdx.x*256 + threadIdx.x;
  if (i >= n) return;                 // n = 44*512, input (44,7,7,512)
  int img = i >> 9, c = i & 511;
  float a = 0.f;
  const h16* p = in + (size_t)img*49*512 + c;
  for (int j = 0; j < 49; ++j) a += (float)p[(size_t)j*512];
  out[i] = (h16)(a * (1.f/49.f));
}

// ======================================================================
// LayerNorm (one block per row)
// ======================================================================
__global__ __launch_bounds__(256) void k_layernorm(const float* __restrict__ x,
                                                   const float* __restrict__ g,
                                                   const float* __restrict__ bta,
                                                   float* outf, h16* outh, int D, int relu){
  __shared__ float s1[256], s2[256];
  int row = blockIdx.x;
  const float* xr = x + (size_t)row*D;
  float a = 0.f, b = 0.f;
  for (int d = threadIdx.x; d < D; d += 256){ float v = xr[d]; a += v; b += v*v; }
  s1[threadIdx.x] = a; s2[threadIdx.x] = b;
  __syncthreads();
  for (int st = 128; st > 0; st >>= 1){
    if (threadIdx.x < st){ s1[threadIdx.x] += s1[threadIdx.x+st]; s2[threadIdx.x] += s2[threadIdx.x+st]; }
    __syncthreads();
  }
  float mean = s1[0] / (float)D;
  float var  = s2[0] / (float)D - mean*mean;
  float inv  = rsqrtf(var + 1e-5f);
  for (int d = threadIdx.x; d < D; d += 256){
    float v = (xr[d] - mean)*inv*g[d] + bta[d];
    if (relu) v = fmaxf(v, 0.f);
    if (outf) outf[(size_t)row*D + d] = v;
    if (outh) outh[(size_t)row*D + d] = (h16)v;
  }
}

// ======================================================================
// Tiny MHA (B=4,N=10,H=8,d=64)
// ======================================================================
__global__ __launch_bounds__(32) void k_attn(const float* __restrict__ qkv, float* __restrict__ o){
  int b = blockIdx.x >> 3, h = blockIdx.x & 7;
  int t = threadIdx.x;
  if (t >= 10) return;
  const float* q = qkv + (size_t)(b*10 + t)*1536 + h*64;
  float s[10], mx = -1e30f;
  for (int j = 0; j < 10; ++j){
    const float* kp = qkv + (size_t)(b*10 + j)*1536 + 512 + h*64;
    float d = 0.f;
    for (int e = 0; e < 64; ++e) d += q[e]*kp[e];
    s[j] = d * 0.125f;
    mx = fmaxf(mx, s[j]);
  }
  float sum = 0.f;
  for (int j = 0; j < 10; ++j){ s[j] = __expf(s[j] - mx); sum += s[j]; }
  float inv = 1.f / sum;
  float acc[64];
  for (int e = 0; e < 64; ++e) acc[e] = 0.f;
  for (int j = 0; j < 10; ++j){
    const float* vp = qkv + (size_t)(b*10 + j)*1536 + 1024 + h*64;
    float w = s[j]*inv;
    for (int e = 0; e < 64; ++e) acc[e] += w*vp[e];
  }
  float* op = o + (size_t)(b*10 + t)*512 + h*64;
  for (int e = 0; e < 64; ++e) op[e] = acc[e];
}

// ======================================================================
// Edge features
// ======================================================================
__global__ __launch_bounds__(256) void k_overlap(const float* __restrict__ obj,
                                                 const float* __restrict__ tgt,
                                                 float* __restrict__ rel){
  __shared__ float s[256];
  int bn = blockIdx.x; int b = bn / 10;
  const float* o = obj + (size_t)bn*50176;
  const float* t = tgt + (size_t)b*50176;
  float a = 0.f;
  for (int p = threadIdx.x; p < 50176; p += 256) a += o[p]*t[p];
  s[threadIdx.x] = a; __syncthreads();
  for (int st = 128; st > 0; st >>= 1){
    if (threadIdx.x < st) s[threadIdx.x] += s[threadIdx.x+st];
    __syncthreads();
  }
  if (threadIdx.x == 0) rel[(size_t)bn*2] = s[0];
}
__global__ __launch_bounds__(64) void k_rowsum(const float* __restrict__ tgt, float* __restrict__ rs){
  __shared__ float s[64];
  int row = blockIdx.x;
  const float* p = tgt + (size_t)row*224;
  float a = 0.f;
  for (int w = threadIdx.x; w < 224; w += 64) a += p[w];
  s[threadIdx.x] = a; __syncthreads();
  for (int st = 32; st > 0; st >>= 1){
    if (threadIdx.x < st) s[threadIdx.x] += s[threadIdx.x+st];
    __syncthreads();
  }
  if (threadIdx.x == 0) rs[row] = s[0];
}
__global__ __launch_bounds__(256) void k_rowtot(const float* __restrict__ rs, float* __restrict__ tot){
  __shared__ float s[256];
  int b = blockIdx.x;
  float a = 0.f;
  for (int h = threadIdx.x; h < 224; h += 256) a += rs[b*224 + h];
  s[threadIdx.x] = a; __syncthreads();
  for (int st = 128; st > 0; st >>= 1){
    if (threadIdx.x < st) s[threadIdx.x] += s[threadIdx.x+st];
    __syncthreads();
  }
  if (threadIdx.x == 0) tot[b] = s[0];
}
__global__ __launch_bounds__(64) void k_iou(const int* __restrict__ bbox, const float* __restrict__ rs,
                                            const float* __restrict__ tot, float* __restrict__ rel){
  int i = threadIdx.x;
  if (i >= 40) return;
  int b  = i / 10;
  int x1 = bbox[i*4+0], y1 = bbox[i*4+1], x2 = bbox[i*4+2], y2 = bbox[i*4+3];
  float has0 = (y1 <= 0 && y2 > 0) ? 1.f : 0.f;
  int lo = x1 < 0 ? 0 : x1;
  int hi = x2 > 224 ? 224 : x2;
  float inter = 0.f;
  for (int h = lo; h < hi; ++h) inter += rs[b*224 + h];
  inter *= has0;
  float cnt = has0 * (float)(hi > lo ? hi - lo : 0);
  float uni = cnt*224.f + tot[b] - inter;
  rel[(size_t)i*2 + 1] = inter / (uni + 1e-8f);
}

// Concat [x(40,512) | spatial(40,512)] f32 -> (4,10240) f16
__global__ __launch_bounds__(256) void k_concat(const float* __restrict__ x, const float* __restrict__ sp,
                                                h16* __restrict__ out, int total){
  int i = blockIdx.x*256 + threadIdx.x;
  if (i >= total) return;
  int row = i >> 10, c = i & 1023;
  float v = (c < 512) ? x[(size_t)row*512 + c] : sp[(size_t)row*512 + (c - 512)];
  out[i] = (h16)v;
}

// ======================================================================
// Host orchestration
// ======================================================================
extern "C" void kernel_launch(void* const* d_in, const int* in_sizes, int n_in,
                              void* d_out, int out_size, void* d_ws, size_t ws_size,
                              hipStream_t stream)
{
  (void)in_sizes; (void)n_in; (void)out_size; (void)ws_size;
  const float* tgt  = (const float*)d_in[0];
  const float* obj  = (const float*)d_in[1];
  const int*   bbox = (const int*)d_in[2];
  auto F = [&](int i)->const float* { return (const float*)d_in[i]; };

  const int P = 3;
  const int LBASE  = P + 0;
  const int OBASE  = P + 72;
  const int OPBASE = P + 78;
  const int RBASE  = P + 92;
  const int WOBASE = P + 154;
  const int WTBASE = P + 164;

  char* wsb = (char*)d_ws;
  size_t off = 0;
  auto alloc = [&](size_t bytes)->void* {
    void* p = wsb + off;
    off = (off + bytes + 255) & ~(size_t)255;
    return p;
  };
  auto allocH = [&](size_t n)->h16*   { return (h16*)  alloc(n*sizeof(h16)); };
  auto allocF = [&](size_t n)->float* { return (float*)alloc(n*sizeof(float)); };

  // activations
  h16* x_in = allocH((size_t)44*224*224);
  h16* bufA = allocH((size_t)44*112*112*64);
  h16* bufB = allocH((size_t)44*56*56*64);
  h16* bufC = allocH((size_t)44*56*56*64);
  h16* bufD = allocH((size_t)44*28*28*128);
  h16* pooled = allocH(44*512);
  float* feats = allocF(44*1024);
  h16* featsH  = allocH(44*1024);
  float* rel = allocF(80);
  h16* relH  = allocH(80);
  float* rsum = allocF(4*224);
  float* rtot = allocF(4);
  float* t0 = allocF(4*1024);  h16* t0h = allocH(4*1024);
  float* spatialF = allocF(4*5120); h16* spatialH = allocH(4*5120);
  float* u0 = allocF(4*2048);  h16* u0h = allocH(4*2048);
  float* u1 = allocF(4*1024);  h16* u1h = allocH(4*1024);
  float* queryF = allocF(4*5120); h16* queryH = allocH(4*5120);
  float* keyF   = allocF(4*5120); h16* keyH   = allocH(4*5120);
  float* xF  = allocF(40*512);
  float* qkv = allocF(40*1536);
  float* attnO = allocF(40*512); h16* attnOH = allocH(40*512);
  float* mhaO  = allocF(40*512);
  h16* x2h = allocH(40*512);
  float* ff1 = allocF(40*2048); h16* ff1h = allocH(40*2048);
  float* ff2 = allocF(40*512);
  h16* combH = allocH(4*10240);
  float* v0 = allocF(4*1024); h16* v0h = allocH(4*1024);
  float* v1 = allocF(4*2048); h16* v1h = allocH(4*2048);
  float* v2 = allocF(4*512);  h16* v2h = allocH(4*512);

  // fp16 weights (linear ones TRANSPOSED [N][K])
  h16* wConv1 = allocH(49*64);
  h16* fcwT   = allocH((size_t)1024*512);
  h16* orl1H  = allocH((size_t)20*1024);
  h16* orl2T  = allocH((size_t)5120*1024);
  h16* wt1T = allocH((size_t)2048*1024);
  h16* wt2T = allocH((size_t)1024*2048);
  h16* wt3T = allocH((size_t)5120*1024);
  h16* wo1T = allocH((size_t)2048*10240);
  h16* wo2T = allocH((size_t)1024*2048);
  h16* wo3T = allocH((size_t)5120*1024);
  h16* op1T = allocH((size_t)1024*10240);
  h16* op2T = allocH((size_t)2048*1024);
  h16* op3T = allocH((size_t)512*2048);
  h16* op4T = allocH((size_t)10*512);
  h16 *LinT[6], *LoutT[6], *Lff1T[6], *Lff2T[6];
  for (int l = 0; l < 6; ++l){
    LinT[l]  = allocH((size_t)1536*512);
    LoutT[l] = allocH((size_t)512*512);
    Lff1T[l] = allocH((size_t)2048*512);
    Lff2T[l] = allocH((size_t)512*2048);
  }

  auto grid1 = [](int n){ return dim3((unsigned)((n + 255)/256)); };
  auto cvt = [&](const float* s, h16* d, int n){
    k_cvt_h<<<grid1(n), dim3(256), 0, stream>>>(s, d, n);
  };
  auto cvtT = [&](const float* s, h16* d, int K, int N){
    k_cvt_lin_t<<<grid1(K*N), dim3(256), 0, stream>>>(s, d, K, N);
  };
  auto cvtconvT = [&](int idx, h16* dst, int Cout,int Cin,int R,int S){
    int n = Cout*Cin*R*S;
    k_cvt_convw_t<<<grid1(n), dim3(256), 0, stream>>>(F(idx), dst, Cout, Cin, R, S);
  };
  auto convF = [&](const h16* in, const h16* wT, const float* bns, const float* bnb,
                   const h16* res, h16* out, int Hin,int Win,int Cin,
                   int Hout,int Wout,int Cout,int R,int S,int st,int pad,int relu){
    int Mtot = 44*Hout*Wout;
    dim3 g((unsigned)((Mtot + 127)/128), (unsigned)(Cout/64));
    k_conv_wmma_c32<<<g, dim3(256), 0, stream>>>(in, wT, bns, bnb, res, out,
                                                 Hin, Win, Cin, Hout, Wout, Cout,
                                                 R, S, st, pad, Mtot, relu);
  };
  auto gemmT = [&](const h16* A, const h16* Bt, const float* bias,
                   float* C, int ldc, int M, int Nn, int K, int relu){
    dim3 g((unsigned)((M + 63)/64), (unsigned)((Nn + 63)/64));
    k_gemm_wmma_t<<<g, dim3(256), 0, stream>>>(A, Bt, bias, C, ldc, M, Nn, K, relu);
  };
  auto ln = [&](const float* x, const float* g, const float* b,
                float* outf, h16* outh, int M, int D, int relu){
    k_layernorm<<<dim3((unsigned)M), dim3(256), 0, stream>>>(x, g, b, outf, outh, D, relu);
  };

  // ---- weight conversion ----
  k_fold_conv1<<<grid1(49*64), dim3(256), 0, stream>>>(F(RBASE+2), wConv1);
  cvtT(F(RBASE+4), fcwT, 512, 1024);
  cvt (F(OBASE+0), orl1H, 20*1024);
  cvtT(F(OBASE+2), orl2T, 1024, 5120);
  cvtT(F(WTBASE+0), wt1T, 1024, 2048);
  cvtT(F(WTBASE+2), wt2T, 2048, 1024);
  cvtT(F(WTBASE+4), wt3T, 1024, 5120);
  cvtT(F(WOBASE+0), wo1T, 10240, 2048);
  cvtT(F(WOBASE+2), wo2T, 2048, 1024);
  cvtT(F(WOBASE+4), wo3T, 1024, 5120);
  cvtT(F(OPBASE+0), op1T, 10240, 1024);
  cvtT(F(OPBASE+2), op2T, 1024, 2048);
  cvtT(F(OPBASE+4), op3T, 2048, 512);
  cvtT(F(OPBASE+6), op4T, 512, 10);
  for (int l = 0; l < 6; ++l){
    int Li = LBASE + l*12;
    cvtT(F(Li+5),  LinT[l],  512, 1536);
    cvtT(F(Li+11), LoutT[l], 512, 512);
    cvtT(F(Li+2),  Lff1T[l], 512, 2048);
    cvtT(F(Li+3),  Lff2T[l], 2048, 512);
  }

  // ---- ResNet-18 on 44 images ----
  k_build_input<<<grid1(44*224*224), dim3(256), 0, stream>>>(tgt, obj, x_in, 44*224*224);
  {
    int Mtot = 44*112*112;
    dim3 g((unsigned)((Mtot + 63)/64), 1);
    k_conv_gen<7,7,1><<<g, dim3(256), 0, stream>>>(x_in, wConv1, F(RBASE+1), F(RBASE+0),
                                                   nullptr, bufA, 224,224, 112,112,64, 2,3, Mtot, 1);
  }
  k_maxpool<<<grid1(44*56*56*64), dim3(256), 0, stream>>>(bufA, bufB, 64,112,112,56,56, 44*56*56*64);

  struct Blk { int base; int cin; int cout; int hin; int stride; bool down; };
  const Blk blks[8] = {
    {RBASE+5,  64, 64, 56, 1, false}, {RBASE+11, 64, 64, 56, 1, false},
    {RBASE+17, 64,128, 56, 2, true }, {RBASE+26,128,128, 28, 1, false},
    {RBASE+32,128,256, 28, 2, true }, {RBASE+41,256,256, 14, 1, false},
    {RBASE+47,256,512, 14, 2, true }, {RBASE+56,512,512,  7, 1, false},
  };
  for (int bi = 0; bi < 8; ++bi){
    const Blk& B = blks[bi];
    int Hout = B.hin / B.stride;
    h16* w1 = allocH((size_t)B.cout*B.cin*9);
    h16* w2 = allocH((size_t)B.cout*B.cout*9);
    h16* wd = B.down ? allocH((size_t)B.cout*B.cin) : nullptr;
    cvtconvT(B.base+4, w1, B.cout, B.cin, 3, 3);
    cvtconvT(B.base+5, w2, B.cout, B.cout, 3, 3);
    if (B.down) cvtconvT(B.base+8, wd, B.cout, B.cin, 1, 1);
    if (B.down)
      convF(bufB, wd, F(B.base+7), F(B.base+6), nullptr, bufD,
            B.hin,B.hin,B.cin, Hout,Hout,B.cout, 1,1,B.stride,0, 0);
    convF(bufB, w1, F(B.base+1), F(B.base+0), nullptr, bufC,
          B.hin,B.hin,B.cin, Hout,Hout,B.cout, 3,3,B.stride,1, 1);
    convF(bufC, w2, F(B.base+3), F(B.base+2), B.down ? bufD : bufB, bufB,
          Hout,Hout,B.cout, Hout,Hout,B.cout, 3,3,1,1, 1);
  }

  // avgpool + fc
  k_avgpool<<<grid1(44*512), dim3(256), 0, stream>>>(bufB, pooled, 44*512);
  gemmT(pooled, fcwT, F(RBASE+3), feats, 1024, 44, 1024, 512, 0);
  cvt(feats, featsH, 44*1024);

  // edge features
  k_overlap<<<dim3(40), dim3(256), 0, stream>>>(obj, tgt, rel);
  k_rowsum<<<dim3(4*224), dim3(64), 0, stream>>>(tgt, rsum);
  k_rowtot<<<dim3(4), dim3(256), 0, stream>>>(rsum, rtot);
  k_iou<<<dim3(1), dim3(64), 0, stream>>>(bbox, rsum, rtot, rel);
  cvt(rel, relH, 80);

  // spatial = mlp2(rel)   (K=20 -> generic gemm)
  k_gemm_wmma<<<dim3(1, 16), dim3(256), 0, stream>>>(relH, orl1H, 1024, F(OBASE+1), t0, 1024, 4, 1024, 20, 0);
  ln(t0, F(OBASE+4), F(OBASE+5), nullptr, t0h, 4, 1024, 1);
  gemmT(t0h, orl2T, F(OBASE+3), spatialF, 5120, 4, 5120, 1024, 0);
  cvt(spatialF, spatialH, 20480);

  // query = mlp3(target_feat) via w_t
  gemmT(featsH, wt1T, F(WTBASE+1), u0, 2048, 4, 2048, 1024, 0);
  ln(u0, F(WTBASE+6), F(WTBASE+7), nullptr, u0h, 4, 2048, 1);
  gemmT(u0h, wt2T, F(WTBASE+3), u1, 1024, 4, 1024, 2048, 0);
  ln(u1, F(WTBASE+8), F(WTBASE+9), nullptr, u1h, 4, 1024, 1);
  gemmT(u1h, wt3T, F(WTBASE+5), queryF, 5120, 4, 5120, 1024, 0);
  cvt(queryF, queryH, 20480);

  // key_ = mlp3(object_feats) via w_o
  gemmT(featsH + (size_t)4*1024, wo1T, F(WOBASE+1), u0, 2048, 4, 2048, 10240, 0);
  ln(u0, F(WOBASE+6), F(WOBASE+7), nullptr, u0h, 4, 2048, 1);
  gemmT(u0h, wo2T, F(WOBASE+3), u1, 1024, 4, 1024, 2048, 0);
  ln(u1, F(WOBASE+8), F(WOBASE+9), nullptr, u1h, 4, 1024, 1);
  gemmT(u1h, wo3T, F(WOBASE+5), keyF, 5120, 4, 5120, 1024, 0);
  cvt(keyF, keyH, 20480);

  k_copy_f<<<grid1(20480), dim3(256), 0, stream>>>(keyF, xF, 20480);

  // transformer layers
  for (int l = 0; l < 6; ++l){
    int Li = LBASE + l*12;
    gemmT(queryH,   LinT[l] + (size_t)0*512,    F(Li+4) + 0,    qkv + 0,    1536, 40, 512, 512, 0);
    gemmT(keyH,     LinT[l] + (size_t)512*512,  F(Li+4) + 512,  qkv + 512,  1536, 40, 512, 512, 0);
    gemmT(spatialH, LinT[l] + (size_t)1024*512, F(Li+4) + 1024, qkv + 1024, 1536, 40, 512, 512, 0);
    k_attn<<<dim3(32), dim3(32), 0, stream>>>(qkv, attnO);
    cvt(attnO, attnOH, 20480);
    gemmT(attnOH, LoutT[l], F(Li+10), mhaO, 512, 40, 512, 512, 0);
    k_add_f<<<grid1(20480), dim3(256), 0, stream>>>(xF, mhaO, 20480);
    ln(xF, F(Li+6), F(Li+7), nullptr, x2h, 40, 512, 0);
    gemmT(x2h, Lff1T[l], F(Li+0), ff1, 2048, 40, 2048, 512, 1);
    cvt(ff1, ff1h, 40*2048);
    gemmT(ff1h, Lff2T[l], F(Li+1), ff2, 512, 40, 512, 2048, 0);
    k_add_f<<<grid1(20480), dim3(256), 0, stream>>>(xF, ff2, 20480);
    ln(xF, F(Li+8), F(Li+9), xF, nullptr, 40, 512, 0);
  }

  // head
  k_concat<<<grid1(40*1024), dim3(256), 0, stream>>>(xF, spatialF, combH, 40*1024);
  gemmT(combH, op1T, F(OPBASE+1), v0, 1024, 4, 1024, 10240, 0);
  ln(v0, F(OPBASE+8),  F(OPBASE+9),  nullptr, v0h, 4, 1024, 1);
  gemmT(v0h, op2T, F(OPBASE+3), v1, 2048, 4, 2048, 1024, 0);
  ln(v1, F(OPBASE+10), F(OPBASE+11), nullptr, v1h, 4, 2048, 1);
  gemmT(v1h, op3T, F(OPBASE+5), v2, 512, 4, 512, 2048, 0);
  ln(v2, F(OPBASE+12), F(OPBASE+13), nullptr, v2h, 4, 512, 1);
  gemmT(v2h, op4T, F(OPBASE+7), (float*)d_out, 10, 4, 10, 512, 0);
}
